// ProjectedConjugatedCSPNet_40759239639736
// MI455X (gfx1250) — compile-verified, matrix operand order checked by reading, not run on hardware
//
#include <hip/hip_runtime.h>

#define NNODE 10000
#define NEDGE 60000
#define HDIM  512
#define LNUM  3
#define NFRQ  128
#define DIS_D 768
#define KP1   2368   // edge-MLP K padded: 512(ef)+512(hi)+512(hj)+768(dis)+6(lat)+58(zero)
                     // 74 K-steps of 32 -> even, so the 2x-unrolled pipeline has no tail

typedef __attribute__((ext_vector_type(16))) __bf16 v16bf;
typedef __attribute__((ext_vector_type(8)))  float  v8f;
typedef __attribute__((ext_vector_type(4)))  int    v4i;

union Frag { v16bf v; uint4 q[2]; };

__device__ __forceinline__ unsigned short f2bf(float x) {
  unsigned u = __float_as_uint(x);
  u += 0x7FFFu + ((u >> 16) & 1u);           // round-to-nearest-even
  return (unsigned short)(u >> 16);
}

__device__ __forceinline__ float silu_f(float x) {
  return x / (1.f + __expf(-x));
}

// ---------------------------------------------------------------- elementwise
__global__ void zero_f32(float* p, int n) {
  int i = blockIdx.x * 256 + threadIdx.x;
  if (i < n) p[i] = 0.f;
}
__global__ void copy_f32(float* d, const float* s, int n) {
  int i = blockIdx.x * 256 + threadIdx.x;
  if (i < n) d[i] = s[i];
}
__global__ void cvt_bf(unsigned short* d, const float* s, int n) {
  int i = blockIdx.x * 256 + threadIdx.x;
  if (i < n) d[i] = f2bf(s[i]);
}
__global__ void count_k(float* cnt, const int* src, int e) {
  int i = blockIdx.x * 256 + threadIdx.x;
  if (i < e) atomicAdd(cnt + src[i], 1.f);
}
__global__ void invc_k(float* c, int n) {
  int i = blockIdx.x * 256 + threadIdx.x;
  if (i < n) c[i] = 1.f / fmaxf(c[i], 1.f);
}
// sinusoid dis-embedding: dis[e, j]=sin, dis[e, 384+j]=cos with j = s*128+f
__global__ void dis_k(unsigned short* dis, const float* fd, int e) {
  int i = blockIdx.x * 256 + threadIdx.x;
  if (i >= e * 384) return;
  int ei = i / 384, j = i % 384;
  int s = j / NFRQ, f = j % NFRQ;
  float v = fd[ei * 3 + s] * 6.283185307179586f * (float)f;
  float sn, cs;
  __sincosf(v, &sn, &cs);
  dis[(size_t)ei * DIS_D + j]       = f2bf(sn);
  dis[(size_t)ei * DIS_D + 384 + j] = f2bf(cs);
}
// eW1 [L,2310,512] -> col-major bf16 [L,512,2368] with region permutation
__global__ void repack_e1(unsigned short* wt, const float* w) {
  int i = blockIdx.x * 256 + threadIdx.x;
  if (i >= LNUM * HDIM * KP1) return;
  int l = i / (HDIM * KP1), r = i % (HDIM * KP1);
  int n = r / KP1, k = r % KP1;
  int row;
  if      (k < 1536) row = k;                 // ef | hi | hj
  else if (k < 2304) row = 1542 + (k - 1536); // dis (orig rows 1542..2309)
  else if (k < 2310) row = 1536 + (k - 2304); // lat (orig rows 1536..1541)
  else               row = -1;                // zero pad
  float v = (row >= 0) ? w[((size_t)l * 2310 + row) * HDIM + n] : 0.f;
  wt[((size_t)l * HDIM + n) * KP1 + k] = f2bf(v);
}
// generic [L,K,512] -> col-major bf16 [L,512,K]
__global__ void repack_t(unsigned short* wt, const float* w, int K) {
  int i = blockIdx.x * 256 + threadIdx.x;
  if (i >= LNUM * HDIM * K) return;
  int l = i / (HDIM * K), r = i % (HDIM * K);
  int n = r / K, k = r % K;
  wt[((size_t)l * HDIM + n) * K + k] = f2bf(w[((size_t)l * K + k) * HDIM + n]);
}

// ------------------------------------------------------------------ layernorm
__global__ __launch_bounds__(256) void ln_k(const float* h, const float* g,
                                            const float* b, unsigned short* hln,
                                            int n) {
  int w = (blockIdx.x * 256 + threadIdx.x) >> 5;
  int lane = threadIdx.x & 31;
  if (w >= n) return;
  const float* row = h + (size_t)w * HDIM;
  float4 v[4];
  float s = 0.f, ss = 0.f;
#pragma unroll
  for (int j = 0; j < 4; ++j) {
    v[j] = *(const float4*)(row + j * 128 + lane * 4);
    s  += v[j].x + v[j].y + v[j].z + v[j].w;
    ss += v[j].x * v[j].x + v[j].y * v[j].y + v[j].z * v[j].z + v[j].w * v[j].w;
  }
#pragma unroll
  for (int m = 16; m >= 1; m >>= 1) {
    s  += __shfl_xor(s, m, 32);
    ss += __shfl_xor(ss, m, 32);
  }
  float mu = s * (1.f / HDIM);
  float rs = rsqrtf(ss * (1.f / HDIM) - mu * mu + 1e-5f);
#pragma unroll
  for (int j = 0; j < 4; ++j) {
    int base = j * 128 + lane * 4;
    ushort4 o;
    o.x = f2bf((v[j].x - mu) * rs * g[base + 0] + b[base + 0]);
    o.y = f2bf((v[j].y - mu) * rs * g[base + 1] + b[base + 1]);
    o.z = f2bf((v[j].z - mu) * rs * g[base + 2] + b[base + 2]);
    o.w = f2bf((v[j].w - mu) * rs * g[base + 3] + b[base + 3]);
    *(ushort4*)(hln + (size_t)w * HDIM + base) = o;
  }
}

// --------------------------------------------------------------- WMMA GEMM
// AMODE: 0 = virtual edge operand (ef|h_ln[src]|h_ln[dst]|dis|lat|0), K=2368
//        1 = plain bf16 activation [R,512]
//        2 = virtual node operand (h_ln | agg*inv_cnt), K=1024
// EPI:   0 = silu -> bf16 store   1 = silu -> atomicAdd agg[src[row]]
//        2 = silu -> h[row] += out (residual)
struct GP {
  int R, K;
  const unsigned short* Wt;    // [512][K] col-major bf16 weights
  const float* bias;           // [512]
  const unsigned short* Abf;   // mode 1
  const unsigned short* ef;    // mode 0
  const unsigned short* hln;   // mode 0/2
  const unsigned short* dis;   // mode 0
  const float* lat;            // mode 0 (lattices [B,6])
  const int* srcIdx;           // mode 0 + epi 1
  const int* dstIdx;           // mode 0
  const int* e2g;              // mode 0
  const float* agg;            // mode 2
  const float* invc;           // mode 2
  unsigned short* Obf;         // epi 0
  float* aggOut;               // epi 1
  float* hOut;                 // epi 2
};

__device__ __forceinline__ uint4 chunk_edge(const GP& p, int e, int s, int d,
                                            int g, int k) {
  if (k < HDIM)       return *(const uint4*)(p.ef  + (size_t)e * HDIM  + k);
  if (k < 2 * HDIM)   return *(const uint4*)(p.hln + (size_t)s * HDIM  + (k - HDIM));
  if (k < 3 * HDIM)   return *(const uint4*)(p.hln + (size_t)d * HDIM  + (k - 2 * HDIM));
  if (k < 2304)       return *(const uint4*)(p.dis + (size_t)e * DIS_D + (k - 1536));
  alignas(16) unsigned short t[8];
#pragma unroll
  for (int i = 0; i < 8; ++i) {
    int j = k + i - 2304;
    t[i] = (j < 6) ? f2bf(p.lat[g * 6 + j]) : (unsigned short)0;
  }
  return *(const uint4*)t;
}

__device__ __forceinline__ uint4 chunk_node(const GP& p, int n, float ic, int k) {
  if (k < HDIM) return *(const uint4*)(p.hln + (size_t)n * HDIM + k);
  const float* a = p.agg + (size_t)n * HDIM + (k - HDIM);
  float4 x = *(const float4*)a;
  float4 y = *(const float4*)(a + 4);
  alignas(16) unsigned short t[8];
  t[0] = f2bf(x.x * ic); t[1] = f2bf(x.y * ic);
  t[2] = f2bf(x.z * ic); t[3] = f2bf(x.w * ic);
  t[4] = f2bf(y.x * ic); t[5] = f2bf(y.y * ic);
  t[6] = f2bf(y.z * ic); t[7] = f2bf(y.w * ic);
  return *(const uint4*)t;
}

template <int AMODE, int EPI>
__global__ __launch_bounds__(256) void gemm_wmma(GP p) {
  // block tile: 256 rows x 64 cols; 8 waves each own 32 rows x 4 col tiles.
  // Explicit 2x ping-pong pipeline; B tiles staged memory->LDS via the CDNA5
  // async path (no VGPR round-trip, drained by s_wait_asynccnt under compute).
  __shared__ unsigned short Bs[2][64 * 40];  // 40-stride pad: conflict-free reads
  const int tid  = threadIdx.x;
  const int wave = tid >> 5, lane = tid & 31;
  const int half = lane >> 4, ln16 = lane & 15;
  const int rbase = blockIdx.x * 256 + wave * 32;
  const int cbase = blockIdx.y * 64;

  int ar0 = rbase + ln16;
  int ar1 = rbase + 16 + ln16;
  int ar0c = ar0 < p.R ? ar0 : p.R - 1;
  int ar1c = ar1 < p.R ? ar1 : p.R - 1;

  int s0 = 0, d0 = 0, g0 = 0, s1 = 0, d1 = 0, g1 = 0;
  float ic0 = 0.f, ic1 = 0.f;
  if (AMODE == 0) {
    s0 = p.srcIdx[ar0c]; d0 = p.dstIdx[ar0c]; g0 = p.e2g[ar0c];
    s1 = p.srcIdx[ar1c]; d1 = p.dstIdx[ar1c]; g1 = p.e2g[ar1c];
  }
  if (AMODE == 2) { ic0 = p.invc[ar0c]; ic1 = p.invc[ar1c]; }

  v8f acc[8] = {};  // [rowfrag*4 + coltile]

  const int stgRow = tid >> 2, stgOff = (tid & 3) * 8;
  auto stageB = [&](int kb, int buf) {
    // GLOBAL_LOAD_ASYNC_TO_LDS_B128: 16B/lane memory->LDS, tracked by ASYNCcnt
    const unsigned short* src = p.Wt + (size_t)(cbase + stgRow) * p.K + kb + stgOff;
    unsigned short* dst = &Bs[buf][stgRow * 40 + stgOff];
    __builtin_amdgcn_global_load_async_to_lds_b128(
        (__attribute__((address_space(1))) v4i*)(unsigned long long)src,
        (__attribute__((address_space(3))) v4i*)(unsigned)(unsigned long long)dst,
        /*offset=*/0, /*cpol=*/0);
  };
  auto drain = [&]() { __builtin_amdgcn_s_wait_asynccnt(0); };

  auto loadA = [&](int kb, Frag& a0, Frag& a1) {
    int k0 = kb + half * 8;  // 16-bit A layout: runs at kb(+8) and kb+16(+8)
    if (AMODE == 0) {
      a0.q[0] = chunk_edge(p, ar0c, s0, d0, g0, k0);
      a0.q[1] = chunk_edge(p, ar0c, s0, d0, g0, k0 + 16);
      a1.q[0] = chunk_edge(p, ar1c, s1, d1, g1, k0);
      a1.q[1] = chunk_edge(p, ar1c, s1, d1, g1, k0 + 16);
    } else if (AMODE == 1) {
      const unsigned short* b0 = p.Abf + (size_t)ar0c * HDIM;
      const unsigned short* b1 = p.Abf + (size_t)ar1c * HDIM;
      a0.q[0] = *(const uint4*)(b0 + k0); a0.q[1] = *(const uint4*)(b0 + k0 + 16);
      a1.q[0] = *(const uint4*)(b1 + k0); a1.q[1] = *(const uint4*)(b1 + k0 + 16);
    } else {
      a0.q[0] = chunk_node(p, ar0c, ic0, k0);
      a0.q[1] = chunk_node(p, ar0c, ic0, k0 + 16);
      a1.q[0] = chunk_node(p, ar1c, ic1, k0);
      a1.q[1] = chunk_node(p, ar1c, ic1, k0 + 16);
    }
  };
  auto compute = [&](const Frag& x0, const Frag& x1, const unsigned short* base) {
#pragma unroll
    for (int n = 0; n < 4; ++n) {
      Frag b;
      const unsigned short* bp = base + (n * 16 + ln16) * 40 + half * 8;
      b.q[0] = *(const uint4*)bp;
      b.q[1] = *(const uint4*)(bp + 16);
      acc[n] = __builtin_amdgcn_wmma_f32_16x16x32_bf16(
          false, x0.v, false, b.v, (short)0, acc[n], false, false);
      acc[4 + n] = __builtin_amdgcn_wmma_f32_16x16x32_bf16(
          false, x1.v, false, b.v, (short)0, acc[4 + n], false, false);
    }
  };

  Frag aA0, aA1, aB0, aB1;
  stageB(0, 0);
  loadA(0, aA0, aA1);
  drain();
  __syncthreads();

  // p.K/32 is even for all instantiations (74, 16, 32 steps)
  for (int kb = 0; kb < p.K; kb += 64) {
    // phase A: compute (aA, Bs[0]); prefetch kb+32 -> (aB, Bs[1])
    stageB(kb + 32, 1);
    loadA(kb + 32, aB0, aB1);
    compute(aA0, aA1, &Bs[0][0]);
    drain();
    __syncthreads();
    // phase B: compute (aB, Bs[1]); prefetch kb+64 -> (aA, Bs[0])
    const bool hasNext = (kb + 64) < p.K;
    if (hasNext) {
      stageB(kb + 64, 0);
      loadA(kb + 64, aA0, aA1);
    }
    compute(aB0, aB1, &Bs[1][0]);
    drain();
    __syncthreads();
  }

  // epilogue: C layout lanes0-15 M=r, lanes16-31 M=r+8; N = lane&15
#pragma unroll
  for (int rf = 0; rf < 2; ++rf) {
    int rb = rbase + rf * 16;
#pragma unroll
    for (int n = 0; n < 4; ++n) {
      int gcol = cbase + n * 16 + ln16;
      float bv = p.bias[gcol];
#pragma unroll
      for (int r = 0; r < 8; ++r) {
        int grow = rb + half * 8 + r;
        if (grow >= p.R) continue;
        float sv = silu_f(acc[rf * 4 + n][r] + bv);
        if (EPI == 0) {
          p.Obf[(size_t)grow * HDIM + gcol] = f2bf(sv);
        } else if (EPI == 1) {
          int sr = p.srcIdx[grow];
          atomicAdd(p.aggOut + (size_t)sr * HDIM + gcol, sv);
        } else {
          p.hOut[(size_t)grow * HDIM + gcol] += sv;
        }
      }
    }
  }
}

// ------------------------------------------------------------------- driver
extern "C" void kernel_launch(void* const* d_in, const int* in_sizes, int n_in,
                              void* d_out, int out_size, void* d_ws,
                              size_t ws_size, hipStream_t stream) {
  const float* node_features = (const float*)d_in[0];
  const float* lattices      = (const float*)d_in[1];
  const float* frac_diff     = (const float*)d_in[2];
  const float* edge_feats    = (const float*)d_in[3];
  const float* ln_g          = (const float*)d_in[4];
  const float* ln_b          = (const float*)d_in[5];
  const float* eW1           = (const float*)d_in[6];
  const float* eb1           = (const float*)d_in[7];
  const float* eW2           = (const float*)d_in[8];
  const float* eb2           = (const float*)d_in[9];
  const float* nW1           = (const float*)d_in[10];
  const float* nb1           = (const float*)d_in[11];
  const float* nW2           = (const float*)d_in[12];
  const float* nb2           = (const float*)d_in[13];
  const int*   edge_index    = (const int*)d_in[14];
  const int*   e2g           = (const int*)d_in[15];
  const int*   src = edge_index;
  const int*   dst = edge_index + NEDGE;

  char* ws = (char*)d_ws;
  size_t off = 0;
  auto take = [&](size_t bytes) -> char* {
    char* p = ws + off;
    off += (bytes + 255) & ~(size_t)255;
    return p;
  };
  unsigned short* hln   = (unsigned short*)take((size_t)NNODE * HDIM * 2);
  unsigned short* ef_bf = (unsigned short*)take((size_t)NEDGE * HDIM * 2);
  unsigned short* disbf = (unsigned short*)take((size_t)NEDGE * DIS_D * 2);
  unsigned short* WtE1  = (unsigned short*)take((size_t)LNUM * HDIM * KP1 * 2);
  unsigned short* WtE2  = (unsigned short*)take((size_t)LNUM * HDIM * HDIM * 2);
  unsigned short* WtN1  = (unsigned short*)take((size_t)LNUM * HDIM * 1024 * 2);
  unsigned short* WtN2  = (unsigned short*)take((size_t)LNUM * HDIM * HDIM * 2);
  unsigned short* t1    = (unsigned short*)take((size_t)NEDGE * HDIM * 2);
  float*          agg   = (float*)take((size_t)NNODE * HDIM * 4);
  float*          invc  = (float*)take((size_t)NNODE * 4);
  float*          h     = (float*)d_out;  // residual state lives in d_out

  auto g1 = [](long n) { return dim3((unsigned)((n + 255) / 256)); };
  dim3 blk(256);

  // one-time per call: counts, conversions, sinusoids, weight repack
  zero_f32<<<g1(NNODE), blk, 0, stream>>>(invc, NNODE);
  count_k<<<g1(NEDGE), blk, 0, stream>>>(invc, src, NEDGE);
  invc_k<<<g1(NNODE), blk, 0, stream>>>(invc, NNODE);
  copy_f32<<<g1((long)NNODE * HDIM), blk, 0, stream>>>(h, node_features, NNODE * HDIM);
  cvt_bf<<<g1((long)NEDGE * HDIM), blk, 0, stream>>>(ef_bf, edge_feats, NEDGE * HDIM);
  dis_k<<<g1((long)NEDGE * 384), blk, 0, stream>>>(disbf, frac_diff, NEDGE);
  repack_e1<<<g1((long)LNUM * HDIM * KP1), blk, 0, stream>>>(WtE1, eW1);
  repack_t<<<g1((long)LNUM * HDIM * HDIM), blk, 0, stream>>>(WtE2, eW2, HDIM);
  repack_t<<<g1((long)LNUM * HDIM * 1024), blk, 0, stream>>>(WtN1, nW1, 1024);
  repack_t<<<g1((long)LNUM * HDIM * HDIM), blk, 0, stream>>>(WtN2, nW2, HDIM);

  dim3 gridE((NEDGE + 255) / 256, 8);
  dim3 gridN((NNODE + 255) / 256, 8);

  for (int l = 0; l < LNUM; ++l) {
    ln_k<<<g1((long)NNODE * 32), blk, 0, stream>>>(h, ln_g + l * HDIM,
                                                   ln_b + l * HDIM, hln, NNODE);
    zero_f32<<<g1((long)NNODE * HDIM), blk, 0, stream>>>(agg, NNODE * HDIM);

    {  // edge MLP layer 1: virtual-concat A, K=2368
      GP p{};
      p.R = NEDGE; p.K = KP1;
      p.Wt = WtE1 + (size_t)l * HDIM * KP1; p.bias = eb1 + l * HDIM;
      p.ef = ef_bf; p.hln = hln; p.dis = disbf; p.lat = lattices;
      p.srcIdx = src; p.dstIdx = dst; p.e2g = e2g;
      p.Obf = t1;
      gemm_wmma<0, 0><<<gridE, blk, 0, stream>>>(p);
    }
    {  // edge MLP layer 2 + fused scatter-add into agg
      GP p{};
      p.R = NEDGE; p.K = HDIM;
      p.Wt = WtE2 + (size_t)l * HDIM * HDIM; p.bias = eb2 + l * HDIM;
      p.Abf = t1; p.srcIdx = src; p.aggOut = agg;
      gemm_wmma<1, 1><<<gridE, blk, 0, stream>>>(p);
    }
    {  // node MLP layer 1: virtual [h_ln | agg*inv_cnt], K=1024
      GP p{};
      p.R = NNODE; p.K = 1024;
      p.Wt = WtN1 + (size_t)l * HDIM * 1024; p.bias = nb1 + l * HDIM;
      p.hln = hln; p.agg = agg; p.invc = invc;
      p.Obf = t1;  // reuse t1 (only first N rows)
      gemm_wmma<2, 0><<<gridN, blk, 0, stream>>>(p);
    }
    {  // node MLP layer 2 + residual add into h
      GP p{};
      p.R = NNODE; p.K = HDIM;
      p.Wt = WtN2 + (size_t)l * HDIM * HDIM; p.bias = nb2 + l * HDIM;
      p.Abf = t1; p.hOut = h;
      gemm_wmma<1, 2><<<gridN, blk, 0, stream>>>(p);
    }
  }
}
